// InteractorwoLSTM_47175920779271
// MI455X (gfx1250) — compile-verified, matrix operand order checked by reading, not run on hardware
//
#include <hip/hip_runtime.h>
#include <stdint.h>

typedef __attribute__((ext_vector_type(16))) _Float16 v16h;
typedef __attribute__((ext_vector_type(8)))  _Float16 v8h;
typedef __attribute__((ext_vector_type(8)))  float    v8f;

#define D    512
#define NTOK 30
#define TT   128
#define BB   32

#define WMMA_F16(a, b, c) __builtin_amdgcn_wmma_f32_16x16x32_f16(false, (a), false, (b), \
                                                                 (short)0, (c), false, false)

// ---------------------------------------------------------------------------
// One-time fp32 -> f16 pack (row-major, size multiple of 2048 elements).
// ---------------------------------------------------------------------------
__global__ __launch_bounds__(256) void cvt_f16(const float* __restrict__ in,
                                               _Float16* __restrict__ outp)
{
    const size_t i = ((size_t)blockIdx.x * 256 + threadIdx.x) * 8;
    const float4 f0 = *(const float4*)(in + i);
    const float4 f1 = *(const float4*)(in + i + 4);
    v8h h;
    h[0] = (_Float16)f0.x; h[1] = (_Float16)f0.y;
    h[2] = (_Float16)f0.z; h[3] = (_Float16)f0.w;
    h[4] = (_Float16)f1.x; h[5] = (_Float16)f1.y;
    h[6] = (_Float16)f1.z; h[7] = (_Float16)f1.w;
    *(v8h*)(outp + i) = h;
}

// ---------------------------------------------------------------------------
// One-time W (512x512, row-major k,n) -> f16 transposed Wt[n*512 + k].
// 32x32 LDS tiles, 256 threads, grid = 16*16 blocks.
// ---------------------------------------------------------------------------
__global__ __launch_bounds__(256) void transpose_cvt_f16(const float* __restrict__ W,
                                                         _Float16* __restrict__ Wt)
{
    __shared__ _Float16 tile[32][33];
    const int bx = blockIdx.x & 15;        // k-tile
    const int by = blockIdx.x >> 4;        // n-tile
    const int c  = threadIdx.x & 31;
    const int r0 = (threadIdx.x >> 5) * 4;
#pragma unroll
    for (int rr = 0; rr < 4; ++rr) {
        const int r = r0 + rr;
        tile[r][c] = (_Float16)W[(size_t)(bx * 32 + r) * D + by * 32 + c];
    }
    __syncthreads();
#pragma unroll
    for (int rr = 0; rr < 4; ++rr) {
        const int r = r0 + rr;
        Wt[(size_t)(by * 32 + r) * D + bx * 32 + c] = tile[c][r];
    }
}

// ---------------------------------------------------------------------------
// C[M x 512] = A[M x 512] * W + bias.  A: f16 row-major, Wt: f16 col-major.
// One wave computes a 32x32 C tile (2 A-frags x 2 B-frags -> 4 WMMAs per
// K-step, 2 b128 loads per WMMA).  Grid = M/16 blocks of 256 threads.
// ---------------------------------------------------------------------------
__global__ __launch_bounds__(256) void gemm_bias_wmma(const _Float16* __restrict__ A,
                                                      const _Float16* __restrict__ Wt,
                                                      const float* __restrict__ bias,
                                                      float* __restrict__ C)
{
    const int lane  = threadIdx.x & 31;
    const int wave  = threadIdx.x >> 5;
    const int tile  = blockIdx.x * 8 + wave;   // 32x32 tile id, N fastest
    const int tm    = tile >> 4;
    const int tn    = tile & 15;
    const int l15   = lane & 15;
    const int khalf = (lane >> 4) << 4;        // lanes 0-15: K 0-15, 16-31: K 16-31

    const _Float16* ap0 = A  + (size_t)(tm * 32 + l15)      * D + khalf;
    const _Float16* ap1 = A  + (size_t)(tm * 32 + 16 + l15) * D + khalf;
    const _Float16* bp0 = Wt + (size_t)(tn * 32 + l15)      * D + khalf;
    const _Float16* bp1 = Wt + (size_t)(tn * 32 + 16 + l15) * D + khalf;

    v8f acc00 = {}, acc01 = {}, acc10 = {}, acc11 = {};
#pragma unroll 2
    for (int k0 = 0; k0 < D; k0 += 32) {
        const v16h a0 = *(const v16h*)(ap0 + k0);
        const v16h a1 = *(const v16h*)(ap1 + k0);
        const v16h b0 = *(const v16h*)(bp0 + k0);
        const v16h b1 = *(const v16h*)(bp1 + k0);
        if (k0 + 32 < D) {
            __builtin_prefetch(ap0 + k0 + 32, 0, 1);   // global_prefetch_b8
            __builtin_prefetch(ap1 + k0 + 32, 0, 1);
        }
        acc00 = WMMA_F16(a0, b0, acc00);               // independent accumulators:
        acc01 = WMMA_F16(a0, b1, acc01);               // no WMMA->WMMA RAW hazard
        acc10 = WMMA_F16(a1, b0, acc10);
        acc11 = WMMA_F16(a1, b1, acc11);
    }

    // C/D layout: VGPR r -> M = base + r + (lane>=16 ? 8 : 0), N = base + lane%16
    const int crow0 = tm * 32 + ((lane >> 4) << 3);
    const int crow1 = crow0 + 16;
    const int ccol0 = tn * 32 + l15;
    const int ccol1 = ccol0 + 16;
    const float bia0 = bias[ccol0];
    const float bia1 = bias[ccol1];
#pragma unroll
    for (int r = 0; r < 8; ++r) {
        C[(size_t)(crow0 + r) * D + ccol0] = acc00[r] + bia0;
        C[(size_t)(crow0 + r) * D + ccol1] = acc01[r] + bia1;
        C[(size_t)(crow1 + r) * D + ccol0] = acc10[r] + bia0;
        C[(size_t)(crow1 + r) * D + ccol1] = acc11[r] + bia1;
    }
}

// ---------------------------------------------------------------------------
// 4 hardware tanh in one block: back-to-back TRANS ops cover each other's
// result-use hazard; one trailing v_nop covers the last (ISA 7.4 rule).
// Early-clobber outputs: block writes before reading all inputs.
// ---------------------------------------------------------------------------
__device__ __forceinline__ void tanh4(float x0, float x1, float x2, float x3,
                                      float& t0, float& t1, float& t2, float& t3)
{
    asm volatile("v_tanh_f32 %0, %4\n\t"
                 "v_tanh_f32 %1, %5\n\t"
                 "v_tanh_f32 %2, %6\n\t"
                 "v_tanh_f32 %3, %7\n\t"
                 "v_nop"
                 : "=&v"(t0), "=&v"(t1), "=&v"(t2), "=&v"(t3)
                 : "v"(x0), "v"(x1), "v"(x2), "v"(x3));
}

// ---------------------------------------------------------------------------
// Per (b,t): beta[n] = sum_d tanh(S[b,n,d]+V[b,t,d])*W_w[d] + b_w, masked
// softmax (exact reference semantics), out[b,t,:] = alpha @ h_s[b].
// V row + W_w staged in LDS via GLOBAL_LOAD_ASYNC_TO_LDS_B128 (ASYNCcnt).
// ---------------------------------------------------------------------------
__global__ __launch_bounds__(256) void fused_attn(const float* __restrict__ S,
                                                  const float* __restrict__ Vm,
                                                  const float* __restrict__ h_s,
                                                  const int*   __restrict__ lengths,
                                                  const float* __restrict__ W_w,
                                                  const float* __restrict__ b_w,
                                                  float* __restrict__ out)
{
    __shared__ __align__(16) float v_sh[D];
    __shared__ __align__(16) float w_sh[D];
    __shared__ float beta_sh[32];
    __shared__ float alpha_sh[32];

    const int bt   = blockIdx.x;
    const int b    = bt >> 7;        // / T
    const int tid  = threadIdx.x;
    const int lane = tid & 31;
    const int wave = tid >> 5;

    // Async stage: threads 0..127 -> V[b,t,:], threads 128..255 -> W_w.
    {
        const int q = (tid & 127) * 4;                 // 4 floats = b128 per lane
        const float*   gsrc = (tid < 128) ? (Vm + (size_t)bt * D + q) : (W_w + q);
        float*         ldst = (tid < 128) ? (v_sh + q) : (w_sh + q);
        const uint32_t lofs = (uint32_t)(uintptr_t)ldst;   // LDS offset = addr[31:0]
        asm volatile("global_load_async_to_lds_b128 %0, %1, off"
                     :: "v"(lofs), "v"(gsrc) : "memory");
        asm volatile("s_wait_asynccnt 0x0" ::: "memory");
    }
    __syncthreads();

    // --- beta[n]: one n per wave; lane owns 4 contiguous d per chunk -> all
    // b128 traffic (global + conflict-free LDS); shuffle reduction.
    const float bw0 = b_w[0];
    for (int n = wave; n < NTOK; n += 8) {
        const float* srow = S + ((size_t)b * NTOK + n) * D;
        float acc = 0.f;
#pragma unroll
        for (int c = 0; c < 4; ++c) {
            const int d = (lane << 2) + (c << 7);
            const float4 s4 = *(const float4*)(srow + d);
            const float4 v4 = *(const float4*)(v_sh + d);
            const float4 w4 = *(const float4*)(w_sh + d);
            float t0, t1, t2, t3;
            tanh4(s4.x + v4.x, s4.y + v4.y, s4.z + v4.z, s4.w + v4.w, t0, t1, t2, t3);
            acc += t0 * w4.x + t1 * w4.y + t2 * w4.z + t3 * w4.w;
        }
#pragma unroll
        for (int off = 16; off; off >>= 1)
            acc += __shfl_xor(acc, off, 32);
        if (lane == 0) beta_sh[n] = acc + bw0;
    }
    __syncthreads();

    // --- masked softmax, exactly as reference: softmax(beta*m) * m, renorm ---
    if (wave == 0) {
        const int n = lane;
        const bool valid = n < NTOK;
        const float m = (valid && lengths[b] >= n + 1) ? 1.f : 0.f;
        const float x = valid ? beta_sh[n] * m : 0.f;
        float mx = valid ? x : -3.402823466e38f;
#pragma unroll
        for (int off = 16; off; off >>= 1)
            mx = fmaxf(mx, __shfl_xor(mx, off, 32));
        float e = valid ? __expf(x - mx) : 0.f;
        float se = e;
#pragma unroll
        for (int off = 16; off; off >>= 1)
            se += __shfl_xor(se, off, 32);
        const float p = (e / se) * m;
        float sp = p;
#pragma unroll
        for (int off = 16; off; off >>= 1)
            sp += __shfl_xor(sp, off, 32);
        if (valid) alpha_sh[n] = p / (sp + 1e-13f) + 1e-13f;
    }
    __syncthreads();

    // --- out[b,t,d] = sum_n alpha[n] * h_s[b,n,d]  (h_s is L2-resident) ---
    {
        const int d = tid << 1;                        // 2 floats per thread
        float2 acc = make_float2(0.f, 0.f);
#pragma unroll
        for (int n = 0; n < NTOK; ++n) {
            const float2 hv = *(const float2*)(h_s + ((size_t)b * NTOK + n) * D + d);
            const float  a  = alpha_sh[n];
            acc.x += a * hv.x;
            acc.y += a * hv.y;
        }
        *(float2*)(out + (size_t)bt * D + d) = acc;
    }
}

// ---------------------------------------------------------------------------
extern "C" void kernel_launch(void* const* d_in, const int* in_sizes, int n_in,
                              void* d_out, int out_size, void* d_ws, size_t ws_size,
                              hipStream_t stream)
{
    const float* h_s     = (const float*)d_in[0];   // (32, 30, 512)
    const float* h_v     = (const float*)d_in[1];   // (32, 128, 512)
    const int*   lengths = (const int*)  d_in[2];   // (32,)
    const float* W_S     = (const float*)d_in[3];   // (512, 512)
    const float* b_S     = (const float*)d_in[4];   // (512,)
    const float* W_V     = (const float*)d_in[5];   // (512, 512)
    const float* b_V     = (const float*)d_in[6];   // (512,)
    const float* W_w     = (const float*)d_in[7];   // (512,)
    const float* b_w     = (const float*)d_in[8];   // (1,)
    float*       out     = (float*)d_out;           // (32, 128, 512)

    // ---- scratch layout (~16.5 MB, fully overwritten each call) ----
    _Float16* Ah_s = (_Float16*)d_ws;                        //  960 x 512 f16
    _Float16* Ah_v = Ah_s + (size_t)BB * NTOK * D;           // 4096 x 512 f16
    _Float16* WtS  = Ah_v + (size_t)BB * TT * D;             //  512 x 512 f16
    _Float16* WtV  = WtS  + (size_t)D * D;                   //  512 x 512 f16
    float*    Smat = (float*)(WtV + (size_t)D * D);          //  960 x 512 f32
    float*    Vmat = Smat + (size_t)BB * NTOK * D;           // 4096 x 512 f32

    // ---- one-time packs: fp32 -> f16 (A row-major, W transposed) ----
    cvt_f16<<<(BB * NTOK * D) / (8 * 256), 256, 0, stream>>>(h_s, Ah_s);   // 240
    cvt_f16<<<(BB * TT   * D) / (8 * 256), 256, 0, stream>>>(h_v, Ah_v);   // 1024
    transpose_cvt_f16<<<256, 256, 0, stream>>>(W_S, WtS);
    transpose_cvt_f16<<<256, 256, 0, stream>>>(W_V, WtV);

    // ---- WMMA GEMMs: S = h_s@W_S + b_S, V = h_v@W_V + b_V ----
    gemm_bias_wmma<<<60,  256, 0, stream>>>(Ah_s, WtS, b_S, Smat);   // M=960
    gemm_bias_wmma<<<256, 256, 0, stream>>>(Ah_v, WtV, b_V, Vmat);   // M=4096

    // ---- fused tanh-attention: one block per (b,t) ----
    fused_attn<<<BB * TT, 256, 0, stream>>>(Smat, Vmat, h_s, lengths, W_w, b_w, out);
}
